// MultiHeadAttentionQuantum_65481071396240
// MI455X (gfx1250) — matrix operand order repaired and checked by reading, not compiled, not run on hardware
//
#include <hip/hip_runtime.h>
#include <hip/hip_bf16.h>

#define N_TOK 8192
#define EMB   64
#define QT    2            // query row-tiles (of 16) per wave in the flash kernel
#define WAVES 4            // waves per flash workgroup (share K/V tiles via TDM->LDS)

typedef _Float16 v16h __attribute__((ext_vector_type(16)));
typedef _Float16 v8h  __attribute__((ext_vector_type(8)));
typedef float    v8f  __attribute__((ext_vector_type(8)));
typedef unsigned int u32x4 __attribute__((ext_vector_type(4)));
typedef int          i32x4 __attribute__((ext_vector_type(4)));
typedef int          i32x8 __attribute__((ext_vector_type(8)));

// ---------------------------------------------------------------------------
// WMMA wrapper: D = A(16x32 f16) * B(32x16 f16) + C(16x16 f32)
// ---------------------------------------------------------------------------
__device__ __forceinline__ v8f wmma16(v16h a, v16h b, v8f c) {
  return __builtin_amdgcn_wmma_f32_16x16x32_f16(
      /*neg_a=*/false, a, /*neg_b=*/false, b,
      /*c_mod=*/(short)0, c, /*reuse_a=*/false, /*reuse_b=*/false);
}

// A-fragment (16x32, f16) from a row-major f16 matrix (global or LDS).
// Lane L (row = L&15): halves 0-15 = K (L>>4)*16+0..15
__device__ __forceinline__ v16h load_a_f16(const _Float16* p, int hi) {
  v8h lo = *(const v8h*)(p + hi * 16);
  v8h h8 = *(const v8h*)(p + hi * 16 + 8);
  v16h a;
#pragma unroll
  for (int i = 0; i < 8; ++i) { a[i] = lo[i]; a[i + 8] = h8[i]; }
  return a;
}

// A-fragment built from an f32 row (convert to f16 on the fly).
__device__ __forceinline__ v16h load_a_f32(const float* p, int hi) {
  v16h a;
#pragma unroll
  for (int i = 0; i < 8; ++i) {
    a[i]     = (_Float16)p[hi * 16 + i];
    a[i + 8] = (_Float16)p[hi * 16 + 8 + i];
  }
  return a;
}

// B-fragment (32x16, f16) for GEMM out = X @ W^T : B[k][n] = W[e0+n][d0+k].
__device__ __forceinline__ v16h load_b_w(const float* W, int e0, int d0,
                                         int lx, int hi) {
  const float* p = W + (e0 + lx) * EMB + d0 + hi * 16;
  v16h b;
#pragma unroll
  for (int i = 0; i < 16; ++i) b[i] = (_Float16)p[i];
  return b;
}

// ---------------------------------------------------------------------------
// Tensor Data Mover: DMA a 2-D f16 tile (tile0 x tile1 elements, row stride
// stride0 elements) from global memory into LDS.  D# built per ISA 08 sect. 8.
// Must be executed by one wave; values must be wave-uniform.  TENSORcnt.
// ---------------------------------------------------------------------------
__device__ __forceinline__ void tdm_load_2d_f16(unsigned lds_byte_addr,
                                                const _Float16* gptr,
                                                unsigned tile0, unsigned tile1,
                                                unsigned long long stride0) {
  unsigned long long ga = (unsigned long long)(uintptr_t)gptr;
  u32x4 g0 = {};
  g0[0] = 1u;                                    // count=1, user descriptor
  g0[1] = lds_byte_addr;                         // lds_addr
  g0[2] = (unsigned)(ga & 0xFFFFFFFFu);          // global_addr[31:0]
  g0[3] = (unsigned)((ga >> 32) & 0x01FFFFFFu)   // global_addr[56:32]
        | (2u << 30);                            // type = 2 ("image")
  unsigned tdim0 = tile0, tdim1 = tile1;         // no OOB clipping needed
  i32x8 g1 = {};
  g1[0] = (int)(1u << 16);                       // data_size = 1 -> 2 bytes
  g1[1] = (int)((tdim0 & 0xFFFFu) << 16);        // tensor_dim0[15:0] @ bits63:48
  g1[2] = (int)(((tdim0 >> 16) & 0xFFFFu) |      // tensor_dim0[31:16]
                ((tdim1 & 0xFFFFu) << 16));      // tensor_dim1[15:0]
  g1[3] = (int)(((tdim1 >> 16) & 0xFFFFu) |      // tensor_dim1[31:16]
                ((tile0 & 0xFFFFu) << 16));      // tile_dim0
  g1[4] = (int)(tile1 & 0xFFFFu);                // tile_dim1 (tile_dim2 = 0)
  g1[5] = (int)(stride0 & 0xFFFFFFFFull);        // tensor_dim0_stride[31:0]
  g1[6] = (int)((stride0 >> 32) & 0xFFFFull);    // tensor_dim0_stride[47:32]
  g1[7] = 0;
  i32x4 z4 = {};
#if __clang_major__ >= 23
  i32x8 z8 = {};
  __builtin_amdgcn_tensor_load_to_lds(g0, g1, z4, z4, z8, 0);
#else
  __builtin_amdgcn_tensor_load_to_lds(g0, g1, z4, z4, 0);
#endif
}

// ---------------------------------------------------------------------------
// Kernel 1: q/k/v projections + quantum map.
//   qq, kq: [N,64] f16 row-major   vqT: [64,N] f16 (transposed for PV B-frags)
// ---------------------------------------------------------------------------
__global__ void __launch_bounds__(32)
qkv_quantum_kernel(const float* __restrict__ x,
                   const float* __restrict__ wq, const float* __restrict__ bq,
                   const float* __restrict__ wk, const float* __restrict__ bk,
                   const float* __restrict__ wv, const float* __restrict__ bv,
                   const float* __restrict__ theta,
                   _Float16* __restrict__ qq, _Float16* __restrict__ kq,
                   _Float16* __restrict__ vqT) {
  const int lane = threadIdx.x & 31;
  const int lx = lane & 15, hi = lane >> 4;
  const int n0 = blockIdx.x * 16;

  v16h ax0 = load_a_f32(x + (n0 + lx) * EMB + 0, hi);
  v16h ax1 = load_a_f32(x + (n0 + lx) * EMB + 32, hi);

  const float* Ws[3] = {wq, wk, wv};
  const float* Bb[3] = {bq, bk, bv};

#pragma unroll
  for (int m = 0; m < 3; ++m) {
#pragma unroll
    for (int e0 = 0; e0 < EMB; e0 += 16) {
      v8f acc = {};
      acc = wmma16(ax0, load_b_w(Ws[m], e0, 0, lx, hi), acc);
      acc = wmma16(ax1, load_b_w(Ws[m], e0, 32, lx, hi), acc);
      const int col = e0 + lx;                 // C layout: col = lane&15
      const float bias = Bb[m][col];
      const float ct = __cosf(theta[col]);
      if (m == 2) {
        v8h pk;
#pragma unroll
        for (int g = 0; g < 8; ++g)
          pk[g] = (_Float16)(ct * __cosf(acc[g] + bias));
        *(v8h*)(vqT + (size_t)col * N_TOK + n0 + hi * 8) = pk;
      } else {
        _Float16* dst = (m == 0) ? qq : kq;
#pragma unroll
        for (int g = 0; g < 8; ++g)
          dst[(n0 + g + 8 * hi) * EMB + col] =
              (_Float16)(ct * __cosf(acc[g] + bias));
      }
    }
  }
}

// ---------------------------------------------------------------------------
// Kernel 2: flash attention, 4 waves/WG sharing TDM-staged K/V tiles in LDS.
// Each wave: QT=2 query tiles (32 rows); WG covers 128 query rows.
// Double-buffered K/V tiles; wave 0 drives the Tensor Data Mover.
// ---------------------------------------------------------------------------
__global__ void __launch_bounds__(32 * WAVES)
flash_attn_kernel(const _Float16* __restrict__ qq,
                  const _Float16* __restrict__ kq,
                  const _Float16* __restrict__ vqT,
                  _Float16* __restrict__ ows) {
  __shared__ __align__(32) _Float16 sK[2][32 * EMB];       // K tile: 32 rows x 64
  __shared__ __align__(32) _Float16 sV[2][EMB * 32];       // V tile: 64 e-rows x 32
  __shared__ __align__(32) _Float16 sP[WAVES][QT * 16 * 32];

  const int tid  = threadIdx.x;
  const int wv_id = tid >> 5;
  const int lane = tid & 31;
  const int lx = lane & 15, hi = lane >> 4;
  const bool isw0 = (wv_id == 0);
  const int qbase = blockIdx.x * (WAVES * QT * 16) + wv_id * (QT * 16);
  const float SCALE = 0.35355339059327373f;  // 1/sqrt(D_K=8)

  _Float16* pbuf = &sP[wv_id][0];

  v16h aq[QT][2];
#pragma unroll
  for (int qt = 0; qt < QT; ++qt) {
    aq[qt][0] = load_a_f16(qq + (qbase + qt * 16 + lx) * EMB + 0, hi);
    aq[qt][1] = load_a_f16(qq + (qbase + qt * 16 + lx) * EMB + 32, hi);
  }

  v8f O[QT][4];
  float mrow[QT][8], lrow[QT][8];
#pragma unroll
  for (int qt = 0; qt < QT; ++qt) {
#pragma unroll
    for (int j = 0; j < 4; ++j) { v8f z = {}; O[qt][j] = z; }
#pragma unroll
    for (int g = 0; g < 8; ++g) { mrow[qt][g] = -1.0e30f; lrow[qt][g] = 0.0f; }
  }

  // Prologue: stage tile 0 (K rows 0..31, V cols 0..31) into buffer 0.
  if (isw0) {
    tdm_load_2d_f16((unsigned)(uintptr_t)&sK[0][0], kq, EMB, 32, EMB);
    tdm_load_2d_f16((unsigned)(uintptr_t)&sV[0][0], vqT, 32, EMB, N_TOK);
  }

#pragma unroll 1
  for (int it = 0; it < N_TOK / 32; ++it) {
    const int c0 = it * 32;
    const int p  = it & 1;
    if (isw0) __builtin_amdgcn_s_wait_tensorcnt(0);  // current tiles landed
    __syncthreads();                                  // publish LDS to all waves
    if (isw0 && (c0 + 32) < N_TOK) {                  // prefetch next tiles (async)
      tdm_load_2d_f16((unsigned)(uintptr_t)&sK[p ^ 1][0],
                      kq + (size_t)(c0 + 32) * EMB, EMB, 32, EMB);
      tdm_load_2d_f16((unsigned)(uintptr_t)&sV[p ^ 1][0],
                      vqT + (c0 + 32), 32, EMB, N_TOK);
    }

    // B-frags of K^T from LDS: B[k][n] = sK[ct*16+n][eb+k] (32B/lane ds loads)
    v16h bk2[2][2];
#pragma unroll
    for (int ctile = 0; ctile < 2; ++ctile)
#pragma unroll
      for (int eb = 0; eb < 2; ++eb)
        bk2[ctile][eb] =
            *(const v16h*)(&sK[p][(ctile * 16 + lx) * EMB + eb * 32 + hi * 16]);

    asm volatile("" ::: "memory");  // keep prior-iter P reads before new P writes

#pragma unroll
    for (int qt = 0; qt < QT; ++qt) {
      v8f s0 = {}, s1 = {};
      s0 = wmma16(aq[qt][0], bk2[0][0], s0);
      s0 = wmma16(aq[qt][1], bk2[0][1], s0);
      s1 = wmma16(aq[qt][0], bk2[1][0], s1);
      s1 = wmma16(aq[qt][1], bk2[1][1], s1);
#pragma unroll
      for (int g = 0; g < 8; ++g) {  // row = g + 8*hi, cols across 16 lanes
        float a0 = s0[g] * SCALE, a1 = s1[g] * SCALE;
        float mx = fmaxf(a0, a1);
#pragma unroll
        for (int msk = 1; msk < 16; msk <<= 1)
          mx = fmaxf(mx, __shfl_xor(mx, msk, 32));
        float mnew = fmaxf(mrow[qt][g], mx);
        float alpha = __expf(mrow[qt][g] - mnew);
        float p0 = __expf(a0 - mnew);
        float p1 = __expf(a1 - mnew);
        float rs = p0 + p1;
#pragma unroll
        for (int msk = 1; msk < 16; msk <<= 1)
          rs += __shfl_xor(rs, msk, 32);
        lrow[qt][g] = lrow[qt][g] * alpha + rs;
        mrow[qt][g] = mnew;
#pragma unroll
        for (int j = 0; j < 4; ++j) O[qt][j][g] *= alpha;
        pbuf[qt * 512 + (g + 8 * hi) * 32 + lx]      = (_Float16)p0;
        pbuf[qt * 512 + (g + 8 * hi) * 32 + 16 + lx] = (_Float16)p1;
      }
    }

    asm volatile("s_wait_dscnt 0" ::: "memory");  // P stores -> A-frag loads

    v16h ap[QT];
#pragma unroll
    for (int qt = 0; qt < QT; ++qt)
      ap[qt] = load_a_f16(pbuf + qt * 512 + lx * 32, hi);

#pragma unroll
    for (int j = 0; j < 4; ++j) {
      // B-frag of V from LDS: B[k][n] = sV[(j*16+n)*32 + k]  (32B/lane)
      v16h bv16 = *(const v16h*)(&sV[p][(j * 16 + lx) * 32 + hi * 16]);
#pragma unroll
      for (int qt = 0; qt < QT; ++qt) O[qt][j] = wmma16(ap[qt], bv16, O[qt][j]);
    }
  }

#pragma unroll
  for (int qt = 0; qt < QT; ++qt)
#pragma unroll
    for (int j = 0; j < 4; ++j)
#pragma unroll
      for (int g = 0; g < 8; ++g) {
        float val = O[qt][j][g] / lrow[qt][g];
        ows[(qbase + qt * 16 + g + 8 * hi) * EMB + j * 16 + lx] = (_Float16)val;
      }
}

// ---------------------------------------------------------------------------
// Kernel 3: final projection out @ wo^T + bo -> f32 d_out
// ---------------------------------------------------------------------------
__global__ void __launch_bounds__(32)
out_proj_kernel(const _Float16* __restrict__ ows,
                const float* __restrict__ wo, const float* __restrict__ bo,
                float* __restrict__ out) {
  const int lane = threadIdx.x & 31;
  const int lx = lane & 15, hi = lane >> 4;
  const int n0 = blockIdx.x * 16;

  v16h ao0 = load_a_f16(ows + (n0 + lx) * EMB + 0, hi);
  v16h ao1 = load_a_f16(ows + (n0 + lx) * EMB + 32, hi);

#pragma unroll
  for (int e0 = 0; e0 < EMB; e0 += 16) {
    v8f acc = {};
    acc = wmma16(ao0, load_b_w(wo, e0, 0, lx, hi), acc);
    acc = wmma16(ao1, load_b_w(wo, e0, 32, lx, hi), acc);
    const int col = e0 + lx;
    const float bias = bo[col];
#pragma unroll
    for (int g = 0; g < 8; ++g)
      out[(n0 + g + 8 * hi) * EMB + col] = acc[g] + bias;
  }
}

// ---------------------------------------------------------------------------
extern "C" void kernel_launch(void* const* d_in, const int* in_sizes, int n_in,
                              void* d_out, int out_size, void* d_ws,
                              size_t ws_size, hipStream_t stream) {
  (void)in_sizes; (void)n_in; (void)out_size; (void)ws_size;
  const float* x     = (const float*)d_in[0];
  const float* wq    = (const float*)d_in[1];
  const float* bq    = (const float*)d_in[2];
  const float* wk    = (const float*)d_in[3];
  const float* bk    = (const float*)d_in[4];
  const float* wv    = (const float*)d_in[5];
  const float* bv    = (const float*)d_in[6];
  const float* theta = (const float*)d_in[7];
  const float* wo    = (const float*)d_in[8];
  const float* bo    = (const float*)d_in[9];
  float* out = (float*)d_out;

  // Workspace layout: 4 f16 matrices of N*64 = 1 MB each (4 MB total).
  _Float16* qq  = (_Float16*)d_ws;
  _Float16* kq  = qq  + (size_t)N_TOK * EMB;
  _Float16* vqT = kq  + (size_t)N_TOK * EMB;
  _Float16* ows = vqT + (size_t)N_TOK * EMB;

  qkv_quantum_kernel<<<N_TOK / 16, dim3(32), 0, stream>>>(
      x, wq, bq, wk, bk, wv, bv, theta, qq, kq, vqT);
  flash_attn_kernel<<<N_TOK / (WAVES * QT * 16), dim3(32 * WAVES), 0, stream>>>(
      qq, kq, vqT, ows);
  out_proj_kernel<<<N_TOK / 16, dim3(32), 0, stream>>>(ows, wo, bo, out);
}
